// TOFM_68143951118615
// MI455X (gfx1250) — compile-verified, matrix operand order checked by reading, not run on hardware
//
#include <hip/hip_runtime.h>
#include <hip/hip_bf16.h>
#include <math.h>

#define B_    32
#define S_    256
#define DIN_  49
#define D_    256
#define H_    8
#define L_    4
#define DFF_  1024
#define DK_   32
#define ROWS_ (B_*S_)

typedef __attribute__((ext_vector_type(16))) _Float16 v16h;
typedef __attribute__((ext_vector_type(8)))  _Float16 v8h;
typedef __attribute__((ext_vector_type(8)))  float    v8f;

enum { MODE_F16 = 0, MODE_QK = 1, MODE_V = 2, MODE_PRE = 3, MODE_GELU = 4 };

__device__ __forceinline__ v8f zero_v8f() {
  v8f z;
#pragma unroll
  for (int i = 0; i < 8; i++) z[i] = 0.0f;
  return z;
}

// Load one lane's A/B fragment for v_wmma_f32_16x16x32_f16.
// p = row base + (lane>>4)*8 halves; chunks are [K0..K0+7] and [K0+16..K0+23].
__device__ __forceinline__ v16h load_frag(const _Float16* p) {
  v8h lo = *(const v8h*)p;
  v8h hi = *(const v8h*)(p + 16);
  v16h r;
#pragma unroll
  for (int i = 0; i < 8; i++) { r[i] = lo[i]; r[i + 8] = hi[i]; }
  return r;
}

__device__ __forceinline__ v8f wmma_f16(v16h a, v16h b, v8f c) {
  return __builtin_amdgcn_wmma_f32_16x16x32_f16(false, a, false, b, (short)0, c,
                                                false, false);
}

__device__ __forceinline__ float gelu_f(float x) {
  return 0.5f * x * (1.0f + erff(x * 0.70710678118654752f));
}

// CDNA5 async global->LDS copy (16B per lane), tracked with ASYNCcnt.
__device__ __forceinline__ void async_load_b128(unsigned lds_off, const void* gaddr) {
  asm volatile("global_load_async_to_lds_b128 %0, %1, off"
               :: "v"(lds_off), "v"(gaddr)
               : "memory");
}
#define WAIT_ASYNC(n) asm volatile("s_wait_asynccnt " #n ::: "memory")

// ---------------------------------------------------------------------------
// Small scalar kernels (stats, bias, posenc, input projection, LN, heads)
// ---------------------------------------------------------------------------

__global__ void posenc_kernel(float* __restrict__ pe) {
  int idx = blockIdx.x * 256 + threadIdx.x;      // S_*D_ total
  int s = idx >> 8;
  int d = idx & 255;
  int k = d >> 1;
  float div = expf(-logf(10000.0f) * (2.0f * (float)k) / 256.0f);
  float ang = (float)s * div;
  pe[idx] = (d & 1) ? cosf(ang) : sinf(ang);
}

__global__ void stats_kernel(const float* __restrict__ x,
                             float* __restrict__ ofin, float* __restrict__ lamb,
                             float* __restrict__ regb) {
  int b = blockIdx.x;
  int s = threadIdx.x;   // 256
  __shared__ float sh[S_];
  __shared__ float red[S_];
  __shared__ float medsh;

  const float* xp = x + ((size_t)b * S_ + s) * DIN_;
  float ofi = xp[0], la = xp[5], rv = xp[8];

  // mean of ofi
  red[s] = ofi; __syncthreads();
  for (int st = 128; st > 0; st >>= 1) { if (s < st) red[s] += red[s + st]; __syncthreads(); }
  float mu = red[0] / (float)S_;
  __syncthreads();
  float dx = ofi - mu;
  red[s] = dx * dx; __syncthreads();
  for (int st = 128; st > 0; st >>= 1) { if (s < st) red[s] += red[s + st]; __syncthreads(); }
  float sd = sqrtf(red[0] / (float)(S_ - 1));   // ddof=1
  __syncthreads();

  ofin[b * S_ + s] = dx / (sd + 1e-8f);
  lamb[b * S_ + s] = la;

  // median: stable rank selection, sorted[(S-1)/2] == rank 127
  sh[s] = rv; __syncthreads();
  int rank = 0;
  for (int j = 0; j < S_; j++) {
    float o = sh[j];
    rank += (o < rv) || (o == rv && j < s);
  }
  if (rank == (S_ - 1) / 2) medsh = rv;
  __syncthreads();
  regb[b * S_ + s] = (rv > medsh) ? 1.0f : 0.0f;
}

__global__ void bias_kernel(const float* __restrict__ gamma,
                            const float* __restrict__ ofin,
                            const float* __restrict__ lamb,
                            const float* __restrict__ regb,
                            float* __restrict__ biasb) {
  int i = blockIdx.x;     // query index
  int b = blockIdx.y;
  int j = threadIdx.x;    // key index
  float g0 = gamma[0], g1 = gamma[1], g2 = gamma[2];
  float oi = ofin[b * S_ + i], oj = ofin[b * S_ + j];
  float li = lamb[b * S_ + i], lj = lamb[b * S_ + j];
  float ri = regb[b * S_ + i], rj = regb[b * S_ + j];
  float v = g0 * oi * oj - g1 * fabsf(li - lj) + g2 * ((ri == rj) ? 1.0f : 0.0f);
  biasb[((size_t)b * S_ + i) * S_ + j] = v;
}

__global__ void convT_kernel(const float* __restrict__ W, _Float16* __restrict__ Wt,
                             int Kd, int Nd) {
  int idx = blockIdx.x * 256 + threadIdx.x;
  if (idx >= Kd * Nd) return;
  int k = idx / Nd, n = idx % Nd;
  Wt[(size_t)n * Kd + k] = (_Float16)W[idx];
}

__global__ void in_proj_kernel(const float* __restrict__ x,
                               const float* __restrict__ W,
                               const float* __restrict__ b,
                               float* __restrict__ pre) {
  int row = blockIdx.x;       // 0..ROWS_-1
  int d = threadIdx.x;        // 256
  __shared__ float xr[DIN_];
  if (d < DIN_) xr[d] = x[(size_t)row * DIN_ + d];
  __syncthreads();
  float acc = b[d];
  for (int i = 0; i < DIN_; i++) acc += xr[i] * W[i * D_ + d];
  pre[(size_t)row * D_ + d] = acc;
}

__global__ void ln_kernel(const float* __restrict__ in,
                          const float* __restrict__ g, const float* __restrict__ be,
                          const float* __restrict__ pe,   // S_*D_ or nullptr
                          float* __restrict__ h32, _Float16* __restrict__ h16) {
  int row = blockIdx.x;
  int d = threadIdx.x;
  __shared__ float red[D_];
  float x = in[(size_t)row * D_ + d];
  red[d] = x; __syncthreads();
  for (int st = 128; st > 0; st >>= 1) { if (d < st) red[d] += red[d + st]; __syncthreads(); }
  float mu = red[0] / (float)D_;
  __syncthreads();
  float dx = x - mu;
  red[d] = dx * dx; __syncthreads();
  for (int st = 128; st > 0; st >>= 1) { if (d < st) red[d] += red[d + st]; __syncthreads(); }
  float var = red[0] / (float)D_;
  float y = dx * rsqrtf(var + 1e-5f) * g[d] + be[d];
  if (pe) y += pe[(row & (S_ - 1)) * D_ + d];
  h32[(size_t)row * D_ + d] = y;
  h16[(size_t)row * D_ + d] = (_Float16)y;
}

__global__ void head_kernel(const float* __restrict__ h32,
                            const float* __restrict__ Wc1, const float* __restrict__ bc1,
                            const float* __restrict__ Wc2, const float* __restrict__ bc2,
                            const float* __restrict__ Wa, const float* __restrict__ ba,
                            float* __restrict__ out) {
  int b = blockIdx.x;
  int t = threadIdx.x;   // 128
  __shared__ float z[D_];
  __shared__ float c[D_ / 2];
  const float* zp = h32 + ((size_t)b * S_ + (S_ - 1)) * D_;
  z[t] = zp[t];
  z[t + 128] = zp[t + 128];
  __syncthreads();
  float a = bc1[t];
  for (int d = 0; d < D_; d++) a += z[d] * Wc1[d * (D_ / 2) + t];
  c[t] = gelu_f(a);
  __syncthreads();
  if (t < 3) {
    float s = bc2[t];
    for (int j = 0; j < D_ / 2; j++) s += c[j] * Wc2[j * 3 + t];
    out[b * 3 + t] = s;                     // logits (B,3) flat
  }
  if (t >= 64 && t < 66) {
    int u = t - 64;
    float s = ba[u];
    for (int d = 0; d < D_; d++) s += z[d] * Wa[d * 2 + u];
    out[B_ * 3 + b * 2 + u] = s;            // aux (B,2) after logits
  }
}

// ---------------------------------------------------------------------------
// WMMA GEMM with async double-buffered LDS staging of the weight tile.
// block = 256 threads = 8 waves; block tile = 128 rows x 64 cols.
// All 8 waves share one B tile (cols n0..n0+63) staged chunk-by-chunk via
// global_load_async_to_lds_b128 (ASYNCcnt), overlapping DMA with WMMAs.
// ---------------------------------------------------------------------------
__global__ __launch_bounds__(256) void gemm_wmma(
    const _Float16* __restrict__ A, const _Float16* __restrict__ Bt,
    const float* __restrict__ bias, const float* __restrict__ resid,
    _Float16* __restrict__ out16, float* __restrict__ out32,
    int K, int N, int mode) {
  __shared__ __align__(16) _Float16 bstage[2][64 * 32];   // 2 x 4KB chunks

  const int tid  = threadIdx.x;
  const int lane = tid & 31;
  const int wave = tid >> 5;
  const int hs = lane >> 4;
  const int lm = lane & 15;
  const int m0 = (blockIdx.x * 8 + wave) * 16;
  const int n0 = blockIdx.y * 64;

  // staging role: each thread copies 16 bytes per chunk
  const int srow = tid >> 2;           // 0..63  (weight row n0+srow)
  const int sseg = tid & 3;            // 0..3   (16B segment within 64B row)
  const unsigned ldsA = (unsigned)(uintptr_t)&bstage[0][0] + (unsigned)(srow * 64 + sseg * 16);
  const unsigned ldsB = (unsigned)(uintptr_t)&bstage[1][0] + (unsigned)(srow * 64 + sseg * 16);
  const _Float16* gsrc = Bt + (size_t)(n0 + srow) * K + sseg * 8;

  v8f acc[4];
#pragma unroll
  for (int t = 0; t < 4; t++) acc[t] = zero_v8f();

  const _Float16* arow = A + (size_t)(m0 + lm) * K + hs * 8;
  const int nk = K >> 5;

  // prefetch chunk 0 into buffer 0
  async_load_b128(ldsA, gsrc);

  for (int i = 0; i < nk; i++) {
    if (i + 1 < nk) {
      async_load_b128(((i + 1) & 1) ? ldsB : ldsA, gsrc + (size_t)(i + 1) * 32);
      WAIT_ASYNC(0x1);     // own in-order async loads: chunk i complete
    } else {
      WAIT_ASYNC(0x0);
    }
    __syncthreads();       // chunk i visible to all waves

    const _Float16* bbuf = (i & 1) ? &bstage[1][0] : &bstage[0][0];
    v16h a = load_frag(arow + (size_t)i * 32);
#pragma unroll
    for (int t = 0; t < 4; t++) {
      v16h b = load_frag(bbuf + (t * 16 + lm) * 32 + hs * 8);
      acc[t] = wmma_f16(a, b, acc[t]);
    }
    __syncthreads();       // all reads of buf (i&1) done before it is re-filled
  }

#pragma unroll
  for (int t = 0; t < 4; t++) {
    const int n = n0 + t * 16 + lm;
    const float bv = bias ? bias[n] : 0.0f;
#pragma unroll
    for (int r = 0; r < 8; r++) {
      const int m = m0 + r + 8 * hs;
      float v = acc[t][r] + bv;
      if (mode == MODE_PRE) {
        out32[(size_t)m * N + n] = v + resid[(size_t)m * N + n];
      } else if (mode == MODE_GELU) {
        out16[(size_t)m * N + n] = (_Float16)gelu_f(v);
      } else if (mode == MODE_QK) {
        const int b_ = m / S_, s = m % S_;
        const int h_ = n / DK_, dk = n % DK_;
        out16[(((size_t)b_ * H_ + h_) * S_ + s) * DK_ + dk] = (_Float16)v;
      } else if (mode == MODE_V) {
        const int b_ = m / S_, s = m % S_;
        const int h_ = n / DK_, dk = n % DK_;
        out16[(((size_t)b_ * H_ + h_) * DK_ + dk) * S_ + s] = (_Float16)v;
      } else {
        out16[(size_t)m * N + n] = (_Float16)v;
      }
    }
  }
}

// ---------------------------------------------------------------------------
// Fused attention: per wave: one 16-row q tile of one (b,h).
// scores (16x256) via 16 WMMAs (K=DK=32), bias+softmax in registers,
// probs staged to LDS in A-layout, ctx (16x32) via 16 WMMAs (K=S=256).
// block = 128 threads = 4 waves, 32 KB LDS.
// ---------------------------------------------------------------------------
__global__ __launch_bounds__(128) void attn_kernel(
    const _Float16* __restrict__ q, const _Float16* __restrict__ k,
    const _Float16* __restrict__ vt, const float* __restrict__ biasb,
    _Float16* __restrict__ ctx) {
  __shared__ __align__(16) _Float16 probs[4][16 * S_];
  const int lane = threadIdx.x & 31;
  const int wave = threadIdx.x >> 5;
  const int hs = lane >> 4;
  const int lm = lane & 15;
  const int gid = blockIdx.x * 4 + wave;      // 0 .. B_*H_*(S_/16)-1
  const int mt = gid & (S_ / 16 - 1);
  const int bh = gid >> 4;
  const int b = bh >> 3;
  const int h = bh & (H_ - 1);

  const _Float16* qp = q + ((size_t)bh * S_ + mt * 16 + lm) * DK_ + hs * 8;
  const v16h af = load_frag(qp);
  const v8f z = zero_v8f();

  v8f sc[16];
#pragma unroll
  for (int nt = 0; nt < 16; nt++) {
    const _Float16* kp = k + ((size_t)bh * S_ + nt * 16 + lm) * DK_ + hs * 8;
    sc[nt] = wmma_f16(af, load_frag(kp), z);
  }

  const float scale = 0.17677669529663687f;   // 1/sqrt(DK)
  const float* bb = biasb + (size_t)b * S_ * S_;
#pragma unroll
  for (int nt = 0; nt < 16; nt++) {
    const int col = nt * 16 + lm;
#pragma unroll
    for (int r = 0; r < 8; r++) {
      const int row = mt * 16 + r + 8 * hs;
      sc[nt][r] = sc[nt][r] * scale + bb[(size_t)row * S_ + col];
    }
  }

  // row-wise softmax: row lives in one 16-lane half (N across lanes, tiles in regs)
  _Float16* pw = &probs[wave][0];
#pragma unroll
  for (int r = 0; r < 8; r++) {
    float mx = -3.0e38f;
#pragma unroll
    for (int nt = 0; nt < 16; nt++) mx = fmaxf(mx, sc[nt][r]);
#pragma unroll
    for (int off = 8; off >= 1; off >>= 1) mx = fmaxf(mx, __shfl_xor(mx, off, 16));
    float sum = 0.0f;
#pragma unroll
    for (int nt = 0; nt < 16; nt++) {
      float e = expf(sc[nt][r] - mx);
      sc[nt][r] = e;
      sum += e;
    }
#pragma unroll
    for (int off = 8; off >= 1; off >>= 1) sum += __shfl_xor(sum, off, 16);
    const float inv = 1.0f / sum;
    const int lrow = r + 8 * hs;
#pragma unroll
    for (int nt = 0; nt < 16; nt++)
      pw[lrow * S_ + nt * 16 + lm] = (_Float16)(sc[nt][r] * inv);
  }
  __syncthreads();

  // ctx = probs(16xS) @ V(SxDK); B fragments from v^T (dk-major, s-contiguous)
  v8f ca[2];
  ca[0] = zero_v8f();
  ca[1] = zero_v8f();
  const _Float16* vb = vt + (size_t)bh * DK_ * S_;
  for (int kt = 0; kt < S_; kt += 32) {
    v16h pa = load_frag(pw + lm * S_ + kt + hs * 8);
#pragma unroll
    for (int t = 0; t < 2; t++) {
      const _Float16* vp = vb + (size_t)(t * 16 + lm) * S_ + kt + hs * 8;
      ca[t] = wmma_f16(pa, load_frag(vp), ca[t]);
    }
  }
#pragma unroll
  for (int t = 0; t < 2; t++) {
    const int dk = t * 16 + lm;
#pragma unroll
    for (int r = 0; r < 8; r++) {
      const int s = mt * 16 + r + 8 * hs;
      ctx[((size_t)b * S_ + s) * D_ + h * DK_ + dk] = (_Float16)ca[t][r];
    }
  }
}

// ---------------------------------------------------------------------------
// Host orchestration
// ---------------------------------------------------------------------------
extern "C" void kernel_launch(void* const* d_in, const int* in_sizes, int n_in,
                              void* d_out, int out_size, void* d_ws, size_t ws_size,
                              hipStream_t stream) {
  (void)in_sizes; (void)n_in; (void)out_size; (void)ws_size;
  const float* x     = (const float*)d_in[0];
  const float* gamma = (const float*)d_in[1];
  const float* W_in  = (const float*)d_in[2];
  const float* b_in  = (const float*)d_in[3];
  const float* g_in  = (const float*)d_in[4];
  const float* be_in = (const float*)d_in[5];
  const float* Wq    = (const float*)d_in[6];
  const float* bq    = (const float*)d_in[7];
  const float* Wk    = (const float*)d_in[8];
  const float* bk    = (const float*)d_in[9];
  const float* Wv    = (const float*)d_in[10];
  const float* bv    = (const float*)d_in[11];
  const float* Wo    = (const float*)d_in[12];
  const float* bo    = (const float*)d_in[13];
  const float* g1    = (const float*)d_in[14];
  const float* be1   = (const float*)d_in[15];
  const float* W1    = (const float*)d_in[16];
  const float* b1    = (const float*)d_in[17];
  const float* W2    = (const float*)d_in[18];
  const float* b2    = (const float*)d_in[19];
  const float* g2    = (const float*)d_in[20];
  const float* be2   = (const float*)d_in[21];
  const float* Wc1   = (const float*)d_in[22];
  const float* bc1   = (const float*)d_in[23];
  const float* Wc2   = (const float*)d_in[24];
  const float* bc2   = (const float*)d_in[25];
  const float* Wa    = (const float*)d_in[26];
  const float* ba    = (const float*)d_in[27];

  char* ws = (char*)d_ws;
  size_t off = 0;
  auto alloc = [&](size_t bytes) -> char* {
    char* p = ws + off;
    off += (bytes + 255) & ~(size_t)255;
    return p;
  };
  float*     pe    = (float*)alloc((size_t)S_ * D_ * 4);
  float*     ofin  = (float*)alloc((size_t)B_ * S_ * 4);
  float*     lamb  = (float*)alloc((size_t)B_ * S_ * 4);
  float*     regb  = (float*)alloc((size_t)B_ * S_ * 4);
  float*     biasb = (float*)alloc((size_t)B_ * S_ * S_ * 4);
  float*     pre32 = (float*)alloc((size_t)ROWS_ * D_ * 4);
  float*     h32   = (float*)alloc((size_t)ROWS_ * D_ * 4);
  _Float16*  h16   = (_Float16*)alloc((size_t)ROWS_ * D_ * 2);
  _Float16*  q16   = (_Float16*)alloc((size_t)ROWS_ * D_ * 2);
  _Float16*  k16   = (_Float16*)alloc((size_t)ROWS_ * D_ * 2);
  _Float16*  v16   = (_Float16*)alloc((size_t)ROWS_ * D_ * 2);
  _Float16*  ctx16 = (_Float16*)alloc((size_t)ROWS_ * D_ * 2);
  _Float16*  ffn16 = (_Float16*)alloc((size_t)ROWS_ * DFF_ * 2);
  _Float16*  WqT   = (_Float16*)alloc((size_t)L_ * D_ * D_ * 2);
  _Float16*  WkT   = (_Float16*)alloc((size_t)L_ * D_ * D_ * 2);
  _Float16*  WvT   = (_Float16*)alloc((size_t)L_ * D_ * D_ * 2);
  _Float16*  WoT   = (_Float16*)alloc((size_t)L_ * D_ * D_ * 2);
  _Float16*  W1T   = (_Float16*)alloc((size_t)L_ * D_ * DFF_ * 2);
  _Float16*  W2T   = (_Float16*)alloc((size_t)L_ * DFF_ * D_ * 2);

  // precompute: pos-enc, pairwise-bias inputs, f16 transposed weights
  posenc_kernel<<<(S_ * D_) / 256, 256, 0, stream>>>(pe);
  stats_kernel<<<B_, S_, 0, stream>>>(x, ofin, lamb, regb);
  bias_kernel<<<dim3(S_, B_), S_, 0, stream>>>(gamma, ofin, lamb, regb, biasb);
  for (int i = 0; i < L_; i++) {
    const int nDD = D_ * D_;
    const int nDF = D_ * DFF_;
    convT_kernel<<<(nDD + 255) / 256, 256, 0, stream>>>(Wq + (size_t)i * nDD, WqT + (size_t)i * nDD, D_, D_);
    convT_kernel<<<(nDD + 255) / 256, 256, 0, stream>>>(Wk + (size_t)i * nDD, WkT + (size_t)i * nDD, D_, D_);
    convT_kernel<<<(nDD + 255) / 256, 256, 0, stream>>>(Wv + (size_t)i * nDD, WvT + (size_t)i * nDD, D_, D_);
    convT_kernel<<<(nDD + 255) / 256, 256, 0, stream>>>(Wo + (size_t)i * nDD, WoT + (size_t)i * nDD, D_, D_);
    convT_kernel<<<(nDF + 255) / 256, 256, 0, stream>>>(W1 + (size_t)i * nDF, W1T + (size_t)i * nDF, D_, DFF_);
    convT_kernel<<<(nDF + 255) / 256, 256, 0, stream>>>(W2 + (size_t)i * nDF, W2T + (size_t)i * nDF, DFF_, D_);
  }

  // h = LN(x @ W_in + b_in) + posenc
  in_proj_kernel<<<ROWS_, D_, 0, stream>>>(x, W_in, b_in, pre32);
  ln_kernel<<<ROWS_, D_, 0, stream>>>(pre32, g_in, be_in, pe, h32, h16);

  const dim3 gD(ROWS_ / 128, D_ / 64);     // N = 256
  const dim3 gF(ROWS_ / 128, DFF_ / 64);   // N = 1024
  const int attnBlocks = (B_ * H_ * (S_ / 16)) / 4;

  for (int i = 0; i < L_; i++) {
    const size_t oDD = (size_t)i * D_ * D_;
    const size_t oDF = (size_t)i * D_ * DFF_;
    gemm_wmma<<<gD, 256, 0, stream>>>(h16, WqT + oDD, bq + i * D_, nullptr,
                                      q16, nullptr, D_, D_, MODE_QK);
    gemm_wmma<<<gD, 256, 0, stream>>>(h16, WkT + oDD, bk + i * D_, nullptr,
                                      k16, nullptr, D_, D_, MODE_QK);
    gemm_wmma<<<gD, 256, 0, stream>>>(h16, WvT + oDD, bv + i * D_, nullptr,
                                      v16, nullptr, D_, D_, MODE_V);
    attn_kernel<<<attnBlocks, 128, 0, stream>>>(q16, k16, v16, biasb, ctx16);
    gemm_wmma<<<gD, 256, 0, stream>>>(ctx16, WoT + oDD, bo + i * D_, h32,
                                      nullptr, pre32, D_, D_, MODE_PRE);
    ln_kernel<<<ROWS_, D_, 0, stream>>>(pre32, g1 + i * D_, be1 + i * D_, nullptr, h32, h16);
    gemm_wmma<<<gF, 256, 0, stream>>>(h16, W1T + oDF, b1 + i * DFF_, nullptr,
                                      ffn16, nullptr, D_, DFF_, MODE_GELU);
    gemm_wmma<<<gD, 256, 0, stream>>>(ffn16, W2T + oDF, b2 + i * D_, h32,
                                      nullptr, pre32, DFF_, D_, MODE_PRE);
    ln_kernel<<<ROWS_, D_, 0, stream>>>(pre32, g2 + i * D_, be2 + i * D_, nullptr, h32, h16);
  }

  head_kernel<<<B_, 128, 0, stream>>>(h32, Wc1, bc1, Wc2, bc2, Wa, ba, (float*)d_out);
}
